// TacotronDecoder_49563922596233
// MI455X (gfx1250) — matrix operand order repaired and checked by reading, not compiled, not run on hardware
//
#include <hip/hip_runtime.h>
#include <hip/hip_bf16.h>

// ---------------------------------------------------------------------------
// Tacotron2 decoder for MI455X (gfx1250, wave32, WMMA).
// Persistent-kernel design: 800 serial steps, grid-wide atomic barriers
// between the ~12 dependent stages of each step. All GEMMs use
// v_wmma_f32_16x16x32_bf16 (bf16 operands, f32 accum). Weights are staged
// once to bf16 (~12 MB, L2-resident on the 192MB L2). Activations are
// staged to bf16 by their producing stages so WMMA inner loops are pure
// global_load_b128 + v_wmma (no in-loop cvt).
// ---------------------------------------------------------------------------

typedef __bf16 bf16_t;
typedef __attribute__((ext_vector_type(16))) __bf16 v16bf;
typedef __attribute__((ext_vector_type(8)))  __bf16 v8bf;
typedef __attribute__((ext_vector_type(8)))  float  v8f;

#define B_     64
#define L_     256
#define T_     800
#define NMEL_  80
#define PRE_   256
#define RNN_   512
#define ENC_   512
#define ATT_   128
#define NF_    32
#define KC_    31
#define GATES_ 2048
#define KA_    1280   /* [prenet 256 | ctx 512 | h 512] */
#define KD_    1536   /* [ah 512 | ctx 512 | h 512]    */
#define KP1_   96     /* NMEL padded 80 -> 96          */

#define NBLK_  64
#define NTHR_  256

// ---------------- workspace layout (byte offsets) ----------------
static constexpr size_t al256(size_t x){ return (x + 255) & ~(size_t)255; }
static constexpr size_t OFF_WA   = 0;                                          // bf16 2048x1280
static constexpr size_t OFF_WD   = al256(OFF_WA  + (size_t)GATES_*KA_*2);      // bf16 2048x1536
static constexpr size_t OFF_WP1  = al256(OFF_WD  + (size_t)GATES_*KD_*2);      // bf16 256x96
static constexpr size_t OFF_WP2  = al256(OFF_WP1 + (size_t)PRE_*KP1_*2);       // bf16 256x256
static constexpr size_t OFF_WP3  = al256(OFF_WP2 + (size_t)PRE_*PRE_*2);       // bf16 256x256
static constexpr size_t OFF_WQ   = al256(OFF_WP3 + (size_t)PRE_*PRE_*2);       // bf16 128x512
static constexpr size_t OFF_WM   = al256(OFF_WQ  + (size_t)ATT_*RNN_*2);       // bf16 128x512
static constexpr size_t OFF_WOUT = al256(OFF_WM  + (size_t)ATT_*ENC_*2);       // bf16 80x1024
static constexpr size_t OFF_BA   = al256(OFF_WOUT+ (size_t)NMEL_*1024*2);      // f32 2048
static constexpr size_t OFF_BD   = al256(OFF_BA  + (size_t)GATES_*4);          // f32 2048
static constexpr size_t OFF_PM   = al256(OFF_BD  + (size_t)GATES_*4);          // f32 B*L*ATT
static constexpr size_t OFF_AH   = al256(OFF_PM  + (size_t)B_*L_*ATT_*4);
static constexpr size_t OFF_AC   = al256(OFF_AH  + (size_t)B_*RNN_*4);
static constexpr size_t OFF_DH   = al256(OFF_AC  + (size_t)B_*RNN_*4);
static constexpr size_t OFF_DC   = al256(OFF_DH  + (size_t)B_*RNN_*4);
static constexpr size_t OFF_CTX  = al256(OFF_DC  + (size_t)B_*RNN_*4);
static constexpr size_t OFF_AW   = al256(OFF_CTX + (size_t)B_*ENC_*4);
static constexpr size_t OFF_AWS  = al256(OFF_AW  + (size_t)B_*L_*4);
static constexpr size_t OFF_XA   = al256(OFF_AWS + (size_t)B_*L_*4);           // bf16 64x1280
static constexpr size_t OFF_XD   = al256(OFF_XA  + (size_t)B_*KA_*2);          // bf16 64x1536
static constexpr size_t OFF_XP1  = al256(OFF_XD  + (size_t)B_*KD_*2);          // bf16 64x256
static constexpr size_t OFF_XP2  = al256(OFF_XP1 + (size_t)B_*PRE_*2);         // bf16 64x256
static constexpr size_t OFF_GT   = al256(OFF_XP2 + (size_t)B_*PRE_*2);         // f32 64x2048
static constexpr size_t OFF_Q    = al256(OFF_GT  + (size_t)B_*GATES_*4);
static constexpr size_t OFF_LOC  = al256(OFF_Q   + (size_t)B_*ATT_*4);
static constexpr size_t OFF_EN   = al256(OFF_LOC + (size_t)B_*L_*NF_*4);
static constexpr size_t OFF_BAR  = al256(OFF_EN  + (size_t)B_*L_*4);

// output layout: mel (B,T,80) | gate (B,T) | align (B,T,L)
#define MEL_BASE   ((size_t)0)
#define GATE_BASE  ((size_t)B_*T_*NMEL_)
#define ALIGN_BASE (GATE_BASE + (size_t)B_*T_)

// ---------------- helpers ----------------
__device__ inline bf16_t f2bf(float f){
  unsigned u = __builtin_bit_cast(unsigned, f);
  unsigned r = u + 0x7FFFu + ((u >> 16) & 1u);       // round-to-nearest-even
  unsigned short s = (unsigned short)(r >> 16);
  return __builtin_bit_cast(bf16_t, s);
}
__device__ inline float sigm(float x){ return 1.0f / (1.0f + expf(-x)); }

// A-fragment (16x32 bf16 MxK) from a bf16 row-major matrix: per lane two
// contiguous 16B chunks at k = kb and k = kb+16 (ISA 7.12.2 layout).
__device__ inline v16bf load_a_bf(const bf16_t* __restrict__ X, int ldk,
                                  int mBase, int kBase){
  const int lane = threadIdx.x & 31;
  const int row  = mBase + (lane & 15);
  const int kb   = kBase + ((lane >> 4) << 3);
  const bf16_t* pr = X + (size_t)row * ldk + kb;
  v8bf lo = *(const v8bf*)pr;
  v8bf hi = *(const v8bf*)(pr + 16);
  return __builtin_shufflevector(lo, hi, 0,1,2,3,4,5,6,7,8,9,10,11,12,13,14,15);
}

// A-fragment gathered from an f32 source (cold paths only).
template <class F>
__device__ inline v16bf load_a_f32(F&& elem, int mBase, int kBase){
  const int lane = threadIdx.x & 31;
  const int row  = mBase + (lane & 15);
  const int kb   = kBase + ((lane >> 4) << 3);
  v16bf a;
#pragma unroll
  for (int e = 0; e < 16; ++e){
    const int k = kb + (e & 7) + ((e >> 3) << 4);
    a[e] = f2bf(elem(row, k));
  }
  return a;
}

// B-fragment (32x16 bf16 KxN) from weight W stored (N rows x K cols) row-major:
// per lane two contiguous 16B chunks at k = kb and k = kb+16 (same as A).
__device__ inline v16bf load_b_w(const bf16_t* __restrict__ W, int ldk,
                                 int nBase, int kBase){
  const int lane = threadIdx.x & 31;
  const int n  = nBase + (lane & 15);
  const int k0 = kBase + ((lane >> 4) << 3);
  const bf16_t* pr = W + (size_t)n * ldk + k0;
  v8bf lo = *(const v8bf*)pr;
  v8bf hi = *(const v8bf*)(pr + 16);
  return __builtin_shufflevector(lo, hi, 0,1,2,3,4,5,6,7,8,9,10,11,12,13,14,15);
}

// hot path: bf16 A x bf16 B
__device__ inline v8f gemm_tile_bf(const bf16_t* __restrict__ X, int ldx,
                                   const bf16_t* __restrict__ W, int ldk,
                                   int mBase, int nBase, int K){
  v8f acc = {0.f,0.f,0.f,0.f,0.f,0.f,0.f,0.f};
  for (int kb = 0; kb < K; kb += 32){
    v16bf a = load_a_bf(X, ldx, mBase, kb);
    v16bf b = load_b_w(W, ldk, nBase, kb);
    acc = __builtin_amdgcn_wmma_f32_16x16x32_bf16(false, a, false, b,
                                                  (short)0, acc, false, false);
  }
  return acc;
}

// cold path: f32 gather A x bf16 B
template <class F>
__device__ inline v8f gemm_tile(F&& elem, const bf16_t* __restrict__ W, int ldk,
                                int mBase, int nBase, int K){
  v8f acc = {0.f,0.f,0.f,0.f,0.f,0.f,0.f,0.f};
  for (int kb = 0; kb < K; kb += 32){
    v16bf a = load_a_f32(elem, mBase, kb);
    v16bf b = load_b_w(W, ldk, nBase, kb);
    acc = __builtin_amdgcn_wmma_f32_16x16x32_bf16(false, a, false, b,
                                                  (short)0, acc, false, false);
  }
  return acc;
}

// D-fragment scatter: VGPR r -> M = mBase + r + ((lane>>4)<<3), N = nBase + (lane&15)
template <class WF>
__device__ inline void store_acc(const v8f& acc, int mBase, int nBase, WF&& wr){
  const int lane = threadIdx.x & 31;
  const int n    = nBase + (lane & 15);
  const int mOff = mBase + ((lane >> 4) << 3);
#pragma unroll
  for (int r = 0; r < 8; ++r) wr(mOff + r, n, acc[r]);
}

// grid-wide two-phase barrier (persistent kernel, all 64 blocks resident)
__device__ inline void gsync(unsigned* bar){
  __syncthreads();
  if (threadIdx.x == 0){
    __threadfence();
    unsigned g = __atomic_load_n(bar + 1, __ATOMIC_ACQUIRE);
    if (atomicAdd(bar, 1u) == (unsigned)(NBLK_ - 1)){
      atomicExch(bar, 0u);
      __threadfence();
      atomicAdd(bar + 1, 1u);
    } else {
      while (__atomic_load_n(bar + 1, __ATOMIC_ACQUIRE) == g)
        __builtin_amdgcn_s_sleep(2);
    }
    __threadfence();
  }
  __syncthreads();
}

// ---------------------------------------------------------------------------
// setup: stage weights to bf16, fuse biases, zero state + bf16 staging + barrier
// ---------------------------------------------------------------------------
__global__ void taco_setup_kernel(char* __restrict__ ws,
    const float* __restrict__ Wp1, const float* __restrict__ Wp2, const float* __restrict__ Wp3,
    const float* __restrict__ aWih, const float* __restrict__ aWhh,
    const float* __restrict__ abih, const float* __restrict__ abhh,
    const float* __restrict__ dWih, const float* __restrict__ dWhh,
    const float* __restrict__ dbih, const float* __restrict__ dbhh,
    const float* __restrict__ Wq, const float* __restrict__ Wm,
    const float* __restrict__ Wout)
{
  bf16_t* WA   = (bf16_t*)(ws + OFF_WA);
  bf16_t* WD   = (bf16_t*)(ws + OFF_WD);
  bf16_t* WP1  = (bf16_t*)(ws + OFF_WP1);
  bf16_t* WP2  = (bf16_t*)(ws + OFF_WP2);
  bf16_t* WP3  = (bf16_t*)(ws + OFF_WP3);
  bf16_t* WQb  = (bf16_t*)(ws + OFF_WQ);
  bf16_t* WMb  = (bf16_t*)(ws + OFF_WM);
  bf16_t* WOUTb= (bf16_t*)(ws + OFF_WOUT);
  float*  BA   = (float*)(ws + OFF_BA);
  float*  BD   = (float*)(ws + OFF_BD);
  float*  AH   = (float*)(ws + OFF_AH);
  float*  AC   = (float*)(ws + OFF_AC);
  float*  DH   = (float*)(ws + OFF_DH);
  float*  DC   = (float*)(ws + OFF_DC);
  float*  CTX  = (float*)(ws + OFF_CTX);
  float*  AW   = (float*)(ws + OFF_AW);
  float*  AWS  = (float*)(ws + OFF_AWS);
  unsigned short* XA = (unsigned short*)(ws + OFF_XA);
  unsigned short* XD = (unsigned short*)(ws + OFF_XD);
  unsigned* bar= (unsigned*)(ws + OFF_BAR);

  const int tid = blockIdx.x * blockDim.x + threadIdx.x;
  const int str = gridDim.x * blockDim.x;

  for (int i = tid; i < GATES_*KA_; i += str){
    int n = i / KA_, k = i - n*KA_;
    WA[i] = f2bf(k < 768 ? aWih[(size_t)n*768 + k] : aWhh[(size_t)n*512 + (k-768)]);
  }
  for (int i = tid; i < GATES_*KD_; i += str){
    int n = i / KD_, k = i - n*KD_;
    WD[i] = f2bf(k < 1024 ? dWih[(size_t)n*1024 + k] : dWhh[(size_t)n*512 + (k-1024)]);
  }
  for (int i = tid; i < PRE_*KP1_; i += str){
    int n = i / KP1_, k = i - n*KP1_;
    WP1[i] = f2bf(k < NMEL_ ? Wp1[(size_t)n*NMEL_ + k] : 0.f);
  }
  for (int i = tid; i < PRE_*PRE_; i += str){ WP2[i] = f2bf(Wp2[i]); WP3[i] = f2bf(Wp3[i]); }
  for (int i = tid; i < ATT_*RNN_; i += str){ WQb[i] = f2bf(Wq[i]);  WMb[i] = f2bf(Wm[i]); }
  for (int i = tid; i < NMEL_*1024; i += str) WOUTb[i] = f2bf(Wout[i]);
  for (int i = tid; i < GATES_; i += str){ BA[i] = abih[i] + abhh[i]; BD[i] = dbih[i] + dbhh[i]; }
  for (int i = tid; i < B_*RNN_; i += str){ AH[i]=0.f; AC[i]=0.f; DH[i]=0.f; DC[i]=0.f; CTX[i]=0.f; }
  for (int i = tid; i < B_*L_;   i += str){ AW[i]=0.f; AWS[i]=0.f; }
  for (int i = tid; i < B_*KA_;  i += str) XA[i] = 0;
  for (int i = tid; i < B_*KD_;  i += str) XD[i] = 0;
  if (tid == 0){ bar[0] = 0u; bar[1] = 0u; }
}

// ---------------------------------------------------------------------------
// processed_memory = encoder_out @ Wm^T   (16384 x 128, K=512) — once per call
// ---------------------------------------------------------------------------
__global__ __launch_bounds__(NTHR_) void taco_pm_kernel(const float* __restrict__ enc,
                                                        char* __restrict__ ws)
{
  const bf16_t* WMb = (const bf16_t*)(ws + OFF_WM);
  float* PM = (float*)(ws + OFF_PM);
  const int gw = blockIdx.x * (NTHR_/32) + (threadIdx.x >> 5);   // 2048 waves
  for (int tile = gw; tile < (B_*L_/16) * (ATT_/16); tile += 2048){
    const int nT = tile & 7, mT = tile >> 3;
    const int mBase = mT*16, nBase = nT*16;
    auto elem = [&](int r, int k)->float { return enc[(size_t)r*ENC_ + k]; };
    v8f acc = gemm_tile(elem, WMb, ENC_, mBase, nBase, ENC_);
    store_acc(acc, mBase, nBase, [&](int m, int n, float v){
      PM[(size_t)m*ATT_ + n] = v;
    });
  }
}

// ---------------------------------------------------------------------------
// persistent decoder: 800 steps, 12 grid-synced stages each
// ---------------------------------------------------------------------------
struct PArgs {
  char* ws;
  const float* enc; const int* lengths; const float* mels;
  const float* Wconv; const float* Wloc; const float* vatt;
  const float* bout; const float* Wgate; const float* bgate;
  float* out;
};

__global__ __launch_bounds__(NTHR_, 1) void taco_decoder_kernel(PArgs p)
{
  char* ws = p.ws;
  const bf16_t* WA   = (const bf16_t*)(ws + OFF_WA);
  const bf16_t* WD   = (const bf16_t*)(ws + OFF_WD);
  const bf16_t* WP1  = (const bf16_t*)(ws + OFF_WP1);
  const bf16_t* WP2  = (const bf16_t*)(ws + OFF_WP2);
  const bf16_t* WP3  = (const bf16_t*)(ws + OFF_WP3);
  const bf16_t* WQb  = (const bf16_t*)(ws + OFF_WQ);
  const bf16_t* WOUTb= (const bf16_t*)(ws + OFF_WOUT);
  const float* BA = (const float*)(ws + OFF_BA);
  const float* BD = (const float*)(ws + OFF_BD);
  const float* PM = (const float*)(ws + OFF_PM);
  float* AH  = (float*)(ws + OFF_AH);
  float* AC  = (float*)(ws + OFF_AC);
  float* DH  = (float*)(ws + OFF_DH);
  float* DC  = (float*)(ws + OFF_DC);
  float* CTX = (float*)(ws + OFF_CTX);
  float* AW  = (float*)(ws + OFF_AW);
  float* AWS = (float*)(ws + OFF_AWS);
  bf16_t* XA  = (bf16_t*)(ws + OFF_XA);   // [x3 | ctx | ah]  64 x 1280
  bf16_t* XD  = (bf16_t*)(ws + OFF_XD);   // [ah | ctx | dh]  64 x 1536
  bf16_t* XP1 = (bf16_t*)(ws + OFF_XP1);  // prenet x1, 64 x 256
  bf16_t* XP2 = (bf16_t*)(ws + OFF_XP2);  // prenet x2, 64 x 256
  float* GT  = (float*)(ws + OFF_GT);
  float* Q   = (float*)(ws + OFF_Q);
  float* LOC = (float*)(ws + OFF_LOC);
  float* EN  = (float*)(ws + OFF_EN);
  unsigned* bar = (unsigned*)(ws + OFF_BAR);

  // attention constants live in LDS for the whole kernel (320KB/WGP)
  __shared__ float sWc[2][KC_][NF_];
  __shared__ float sWloc[ATT_][NF_];
  __shared__ float sV[ATT_];
  for (int i = threadIdx.x; i < 2*KC_*NF_; i += NTHR_){
    int f = i % NF_, rem = i / NF_, k = rem % KC_, c = rem / KC_;
    sWc[c][k][f] = p.Wconv[(size_t)f*(2*KC_) + c*KC_ + k];
  }
  for (int i = threadIdx.x; i < ATT_*NF_; i += NTHR_) sWloc[i / NF_][i % NF_] = p.Wloc[i];
  for (int i = threadIdx.x; i < ATT_;     i += NTHR_) sV[i] = p.vatt[i];
  __syncthreads();

  const int gwave = blockIdx.x * (NTHR_/32) + (threadIdx.x >> 5);  // 0..511
  const int lane  = threadIdx.x & 31;
  const int gtid  = blockIdx.x * NTHR_ + threadIdx.x;              // 0..16383

  for (int t = 0; t < T_; ++t){
    // ---- P0: prenet-1 GEMM (waves 0..63) || location conv (blocks 8..63) ----
    if (gwave < 64){
      const int mBase = (gwave & 3) * 16, nBase = (gwave >> 2) * 16;
      auto elem = [&](int r, int k)->float {
        return (t == 0 || k >= NMEL_) ? 0.f
             : p.mels[(size_t)r*T_*NMEL_ + (size_t)(t-1)*NMEL_ + k];
      };
      v8f acc = gemm_tile(elem, WP1, KP1_, mBase, nBase, KP1_);
      store_acc(acc, mBase, nBase, [&](int m, int n, float v){
        XP1[m*PRE_ + n] = f2bf(fmaxf(v, 0.f));
      });
    } else {
      const int cid = (blockIdx.x - 8) * NTHR_ + threadIdx.x;      // 0..14335
      for (int pos = cid; pos < B_*L_; pos += 56*NTHR_){
        const int b = pos >> 8, l = pos & 255;
        float acc[NF_];
#pragma unroll
        for (int f = 0; f < NF_; ++f) acc[f] = 0.f;
#pragma unroll 1
        for (int k = 0; k < KC_; ++k){
          const int ll = l + k - (KC_-1)/2;
          float a0 = 0.f, a1 = 0.f;
          if (ll >= 0 && ll < L_){ a0 = AW[b*L_ + ll]; a1 = AWS[b*L_ + ll]; }
#pragma unroll
          for (int f = 0; f < NF_; ++f) acc[f] += a0*sWc[0][k][f] + a1*sWc[1][k][f];
        }
#pragma unroll
        for (int f = 0; f < NF_; ++f) LOC[(size_t)pos*NF_ + f] = acc[f];
      }
    }
    gsync(bar);

    // ---- P1: prenet-2 (bf16 A) ----
    if (gwave < 64){
      const int mBase = (gwave & 3) * 16, nBase = (gwave >> 2) * 16;
      v8f acc = gemm_tile_bf(XP1, PRE_, WP2, PRE_, mBase, nBase, PRE_);
      store_acc(acc, mBase, nBase, [&](int m, int n, float v){
        XP2[m*PRE_ + n] = f2bf(fmaxf(v, 0.f));
      });
    }
    gsync(bar);

    // ---- P2: prenet-3 -> XA[:,0:256] ----
    if (gwave < 64){
      const int mBase = (gwave & 3) * 16, nBase = (gwave >> 2) * 16;
      v8f acc = gemm_tile_bf(XP2, PRE_, WP3, PRE_, mBase, nBase, PRE_);
      store_acc(acc, mBase, nBase, [&](int m, int n, float v){
        XA[(size_t)m*KA_ + n] = f2bf(fmaxf(v, 0.f));
      });
    }
    gsync(bar);

    // ---- A0: attention-LSTM gates: XA=[x|ctx|h] @ [aWih|aWhh]^T (512 waves) ----
    {
      const int mBase = (gwave & 3) * 16, nBase = (gwave >> 2) * 16;
      v8f acc = gemm_tile_bf(XA, KA_, WA, KA_, mBase, nBase, KA_);
      store_acc(acc, mBase, nBase, [&](int m, int n, float v){
        GT[(size_t)m*GATES_ + n] = v + BA[n];
      });
    }
    gsync(bar);

    // ---- A1: att LSTM pointwise + zoneout; stage ah (bf16) for A0'/A2/D0 ----
    for (int o = gtid; o < B_*RNN_; o += NBLK_*NTHR_){
      const int b = o >> 9, j = o & 511;
      const float* g = GT + (size_t)b*GATES_;
      const float gi = g[j], gf = g[512+j], gc = g[1024+j], go = g[1536+j];
      const float co = AC[o], ho = AH[o];
      const float c2 = sigm(gf)*co + sigm(gi)*tanhf(gc);
      const float h2 = sigm(go)*tanhf(c2);
      const float hz = 0.1f*ho + 0.9f*h2;
      AH[o] = hz;
      AC[o] = 0.1f*co + 0.9f*c2;
      const bf16_t hb = f2bf(hz);
      XA[(size_t)b*KA_ + 768 + j] = hb;   // next step's gate GEMM
      XD[(size_t)b*KD_ + j]       = hb;   // this step's q GEMM + dec gates
    }
    gsync(bar);

    // ---- A2: q = ah2 @ Wq^T (waves 0..31, A = XD[:,0:512]) ----
    if (gwave < 32){
      const int mBase = (gwave & 3) * 16, nBase = (gwave >> 2) * 16;
      v8f acc = gemm_tile_bf(XD, KD_, WQb, RNN_, mBase, nBase, RNN_);
      store_acc(acc, mBase, nBase, [&](int m, int n, float v){ Q[m*ATT_ + n] = v; });
    }
    gsync(bar);

    // ---- A3: energies = v . tanh(q + pm + loc@Wloc^T), masked ----
    {
      const int pos = gtid, b = pos >> 8, l = pos & 255;
      float lb[NF_];
#pragma unroll
      for (int f = 0; f < NF_; ++f) lb[f] = LOC[(size_t)pos*NF_ + f];
      const float* qb  = Q  + b*ATT_;
      const float* pmb = PM + (size_t)pos*ATT_;
      float acc = 0.f;
#pragma unroll 1
      for (int a = 0; a < ATT_; ++a){
        float pl = 0.f;
#pragma unroll
        for (int f = 0; f < NF_; ++f) pl += lb[f] * sWloc[a][f];
        acc += sV[a] * tanhf(qb[a] + pmb[a] + pl);
      }
      if (l >= p.lengths[b]) acc = -1e9f;
      EN[pos] = acc;
    }
    gsync(bar);

    // ---- A4: softmax over L per batch row (wave per row), aws+=, align out ----
    if (gwave < 64){
      const int b = gwave;
      float e[8]; float m = -3.0e38f;
#pragma unroll
      for (int j = 0; j < 8; ++j){ e[j] = EN[b*L_ + j*32 + lane]; m = fmaxf(m, e[j]); }
#pragma unroll
      for (int off = 16; off >= 1; off >>= 1) m = fmaxf(m, __shfl_xor(m, off, 32));
      float s = 0.f;
#pragma unroll
      for (int j = 0; j < 8; ++j){ e[j] = expf(e[j] - m); s += e[j]; }
#pragma unroll
      for (int off = 16; off >= 1; off >>= 1) s += __shfl_xor(s, off, 32);
      const float inv = 1.0f / s;
#pragma unroll
      for (int j = 0; j < 8; ++j){
        const int l = j*32 + lane;
        const float w = e[j] * inv;
        AW[b*L_ + l] = w;
        AWS[b*L_ + l] += w;
        p.out[ALIGN_BASE + (size_t)b*T_*L_ + (size_t)t*L_ + l] = w;
      }
    }
    gsync(bar);

    // ---- A5: ctx = w @ encoder_out; stage ctx (bf16) for D0 and next A0 ----
    for (int o = gtid; o < B_*ENC_; o += NBLK_*NTHR_){
      const int b = o >> 9, e = o & 511;
      const float* w  = AW + b*L_;
      const float* eb = p.enc + (size_t)b*L_*ENC_ + e;
      float s = 0.f;
#pragma unroll 4
      for (int l = 0; l < L_; ++l) s += w[l] * eb[(size_t)l*ENC_];
      CTX[o] = s;
      const bf16_t cb = f2bf(s);
      XD[(size_t)b*KD_ + 512 + e] = cb;   // this step's dec gates
      XA[(size_t)b*KA_ + 256 + e] = cb;   // next step's att gates
    }
    gsync(bar);

    // ---- D0: decoder-LSTM gates: XD=[ah|ctx|dh] @ [dWih|dWhh]^T (512 waves) ----
    {
      const int mBase = (gwave & 3) * 16, nBase = (gwave >> 2) * 16;
      v8f acc = gemm_tile_bf(XD, KD_, WD, KD_, mBase, nBase, KD_);
      store_acc(acc, mBase, nBase, [&](int m, int n, float v){
        GT[(size_t)m*GATES_ + n] = v + BD[n];
      });
    }
    gsync(bar);

    // ---- D1: dec LSTM pointwise + zoneout; stage dh (bf16) for next D0 ----
    for (int o = gtid; o < B_*RNN_; o += NBLK_*NTHR_){
      const int b = o >> 9, j = o & 511;
      const float* g = GT + (size_t)b*GATES_;
      const float gi = g[j], gf = g[512+j], gc = g[1024+j], go = g[1536+j];
      const float co = DC[o], ho = DH[o];
      const float c2 = sigm(gf)*co + sigm(gi)*tanhf(gc);
      const float h2 = sigm(go)*tanhf(c2);
      const float hz = 0.1f*ho + 0.9f*h2;
      DH[o] = hz;
      DC[o] = 0.1f*co + 0.9f*c2;
      XD[(size_t)b*KD_ + 1024 + j] = f2bf(hz);
    }
    gsync(bar);

    // ---- D2: mel = [dh|ctx] @ Wout^T + bout (waves 0..19); gate dot (20..83) ----
    if (gwave < 20){
      const int mBase = (gwave & 3) * 16, nBase = (gwave >> 2) * 16;   // 4 x 5 tiles
      auto elem = [&](int r, int k)->float {
        return (k < 512) ? DH[r*RNN_ + k] : CTX[r*ENC_ + (k-512)];
      };
      v8f acc = gemm_tile(elem, WOUTb, 1024, mBase, nBase, 1024);
      store_acc(acc, mBase, nBase, [&](int m, int n, float v){
        p.out[MEL_BASE + (size_t)m*T_*NMEL_ + (size_t)t*NMEL_ + n] = v + p.bout[n];
      });
    } else if (gwave < 84){
      const int b = gwave - 20;
      float s = 0.f;
#pragma unroll
      for (int i = 0; i < 32; ++i){
        const int k = i*32 + lane;
        const float x = (k < 512) ? DH[b*RNN_ + k] : CTX[b*ENC_ + (k-512)];
        s += x * p.Wgate[k];
      }
#pragma unroll
      for (int off = 16; off >= 1; off >>= 1) s += __shfl_xor(s, off, 32);
      if (lane == 0) p.out[GATE_BASE + (size_t)b*T_ + t] = s + p.bgate[0];
    }
    gsync(bar);
  }
}

// ---------------------------------------------------------------------------
extern "C" void kernel_launch(void* const* d_in, const int* in_sizes, int n_in,
                              void* d_out, int out_size, void* d_ws, size_t ws_size,
                              hipStream_t stream)
{
  (void)in_sizes; (void)n_in; (void)out_size; (void)ws_size;
  const float* enc     = (const float*)d_in[0];
  const int*   lengths = (const int*)  d_in[1];
  const float* mels    = (const float*)d_in[2];
  const float* Wp1  = (const float*)d_in[3];
  const float* Wp2  = (const float*)d_in[4];
  const float* Wp3  = (const float*)d_in[5];
  const float* aWih = (const float*)d_in[6];
  const float* aWhh = (const float*)d_in[7];
  const float* abih = (const float*)d_in[8];
  const float* abhh = (const float*)d_in[9];
  const float* dWih = (const float*)d_in[10];
  const float* dWhh = (const float*)d_in[11];
  const float* dbih = (const float*)d_in[12];
  const float* dbhh = (const float*)d_in[13];
  const float* Wq   = (const float*)d_in[14];
  const float* Wm   = (const float*)d_in[15];
  const float* Wconv= (const float*)d_in[16];
  const float* Wloc = (const float*)d_in[17];
  const float* vatt = (const float*)d_in[18];
  const float* Wout = (const float*)d_in[19];
  const float* bout = (const float*)d_in[20];
  const float* Wgate= (const float*)d_in[21];
  const float* bgate= (const float*)d_in[22];
  float* out = (float*)d_out;
  char*  ws  = (char*)d_ws;

  taco_setup_kernel<<<2048, 256, 0, stream>>>(ws, Wp1, Wp2, Wp3,
      aWih, aWhh, abih, abhh, dWih, dWhh, dbih, dbhh, Wq, Wm, Wout);

  taco_pm_kernel<<<256, NTHR_, 0, stream>>>(enc, ws);

  PArgs p{ws, enc, lengths, mels, Wconv, Wloc, vatt, bout, Wgate, bgate, out};
  taco_decoder_kernel<<<NBLK_, NTHR_, 0, stream>>>(p);
}
